// Iteration_dec_44985487458947
// MI455X (gfx1250) — matrix operand order repaired and checked by reading, not compiled, not run on hardware
//
#include <hip/hip_runtime.h>

#define BATCH 512
#define LEN   1024
#define DIM   64
#define KW    5
#define EPSBN 1e-3f

typedef _Float16 v16h __attribute__((ext_vector_type(16)));
typedef _Float16 v8h  __attribute__((ext_vector_type(8)));
typedef float    v8f  __attribute__((ext_vector_type(8)));

__device__ __forceinline__ float elu1(float y) {
    return y > 0.f ? y : (__expf(y) - 1.f);
}

// One workgroup per sample: whole 5-layer x-branch, dense head, interleave,
// 5-layer I-branch, dense head, de-interleave — activations live in LDS.
__launch_bounds__(256, 1)
__global__ void fused_turbo_decoder(
    const float* __restrict__ dx, const float* __restrict__ de,
    const float* __restrict__ dI, const float* __restrict__ p,
    const float* __restrict__ Wx_first, const float* __restrict__ bx_first,
    const float* __restrict__ Wx_rest,  const float* __restrict__ bx_rest,
    const float* __restrict__ gx, const float* __restrict__ btx,
    const float* __restrict__ mx, const float* __restrict__ vx,
    const float* __restrict__ WI_first, const float* __restrict__ bI_first,
    const float* __restrict__ WI_rest,  const float* __restrict__ bI_rest,
    const float* __restrict__ gI, const float* __restrict__ btI,
    const float* __restrict__ mI, const float* __restrict__ vI,
    const float* __restrict__ fc1_W, const float* __restrict__ fc1_b,
    const float* __restrict__ fc2_W, const float* __restrict__ fc2_b,
    const int* __restrict__ perm, const int* __restrict__ inv_perm,
    float* __restrict__ out)
{
    // 131072 + 131072 + 40960 + 4096 + 4608 = 311808 B of LDS (<= 320 KB/WG)
    __shared__ __align__(16) _Float16 sActA[LEN * DIM];
    __shared__ __align__(16) _Float16 sActB[LEN * DIM];
    __shared__ __align__(16) _Float16 sWswz[10 * 4 * 32 * 16]; // [chunk][ntile][lane][16]
    __shared__ float sDense[LEN];
    __shared__ float sScr[1152];   // [0,960): first-layer W, [960,1024): scale, [1024,1088): shift

    const int b      = blockIdx.x;
    const int tid    = threadIdx.x;
    const int lane   = tid & 31;
    const int wv     = tid >> 5;          // 8 waves
    const int laneHi = lane >> 4;
    const int lane15 = lane & 15;
    const int ntile  = wv & 3;            // output-channel tile of this wave
    const int dch    = ntile * 16 + lane15; // this thread's output channel (WMMA epilogue)
    const int rt0    = (wv >> 2) * 32;    // 32 row-tiles of 16 per wave

    const int* permRow = perm + b * LEN;

    for (int br = 0; br < 2; ++br) {
        const float* Wfirst = br ? WI_first : Wx_first;
        const float* bfirst = br ? bI_first : bx_first;
        const float* Wrest  = br ? WI_rest  : Wx_rest;
        const float* brest  = br ? bI_rest  : bx_rest;
        const float* gg     = br ? gI  : gx;
        const float* bt     = br ? btI : btx;
        const float* mm     = br ? mI  : mx;
        const float* vv     = br ? vI  : vx;
        const float* fcW    = br ? fc2_W : fc1_W;
        const float* fcb    = br ? fc2_b : fc1_b;

        // ---------------- first conv layer (Cin = 3) -> sActA ----------------
        __syncthreads();
        for (int e = tid; e < KW * 3 * DIM; e += 256) sScr[e] = Wfirst[e];
        if (tid < DIM) {
            float s = gg[tid] * rsqrtf(vv[tid] + EPSBN);
            sScr[960  + tid] = s;
            sScr[1024 + tid] = (bfirst[tid] - mm[tid]) * s + bt[tid];
        }
        __syncthreads();

        for (int l = tid; l < LEN; l += 256) {
            float iv[KW][3];
            #pragma unroll
            for (int k = 0; k < KW; ++k) {
                int lp = (l + k + (LEN - 2)) & (LEN - 1);   // circular, pad 2
                if (br == 0) {
                    iv[k][0] = de[b * LEN + lp];
                    iv[k][1] = p [b * LEN + lp];
                    iv[k][2] = dx[b * LEN + lp];
                } else {
                    int pp = permRow[lp];
                    iv[k][0] = dI[b * LEN + lp];
                    iv[k][1] = sDense[pp];                   // interleaved q
                    iv[k][2] = dx[b * LEN + pp];             // interleaved dx
                }
            }
            for (int d = 0; d < DIM; ++d) {
                float acc = 0.f;
                #pragma unroll
                for (int k = 0; k < KW; ++k)
                    #pragma unroll
                    for (int c = 0; c < 3; ++c)
                        acc = fmaf(iv[k][c], sScr[(k * 3 + c) * DIM + d], acc);
                acc = acc * sScr[960 + d] + sScr[1024 + d];
                sActA[l * DIM + d] = (_Float16)elu1(acc);
            }
        }

        // ---------------- 4 WMMA conv layers, ping-pong A<->B ----------------
        for (int j = 0; j < 4; ++j) {
            const _Float16* actIn  = (j & 1) ? sActB : sActA;
            _Float16*       actOut = (j & 1) ? sActA : sActB;
            const float* Wg = Wrest + j * KW * DIM * DIM;   // [k][cin][cout]

            // Swizzle weights into per-fragment layout: B 32x16, 16-bit:
            // lanes 0-15 hold K={0..7,16..23} of column n=lane, lanes 16-31 hold K={8..15,24..31}.
            for (int e = tid; e < 10 * 4 * 32 * 16; e += 256) {
                int i    = e & 15;
                int ln   = (e >> 4) & 31;
                int fr   = e >> 9;
                int ck   = fr >> 2;
                int nt   = fr & 3;
                int kin  = (ln >> 4) * 8 + (i & 7) + (i >> 3) * 16; // K within 32-chunk
                int k    = ck >> 1;
                int cin  = (ck & 1) * 32 + kin;
                int cout = nt * 16 + (ln & 15);
                sWswz[e] = (_Float16)Wg[(k * DIM + cin) * DIM + cout];
            }
            const int li = j + 1;
            float s  = gg[li * DIM + dch] * rsqrtf(vv[li * DIM + dch] + EPSBN);
            float sh = (brest[j * DIM + dch] - mm[li * DIM + dch]) * s + bt[li * DIM + dch];
            __syncthreads();

            // Pin the 10 B fragments of this wave's column tile in VGPRs.
            v16h bfrag[10];
            #pragma unroll
            for (int ck = 0; ck < 10; ++ck)
                bfrag[ck] = *reinterpret_cast<const v16h*>(
                    sWswz + ((ck * 4 + ntile) * 32 + lane) * 16);

            for (int rt = rt0; rt < rt0 + 32; ++rt) {
                const int l0 = rt * 16;
                v8f acc = {0.f, 0.f, 0.f, 0.f, 0.f, 0.f, 0.f, 0.f};
                #pragma unroll
                for (int ck = 0; ck < 10; ++ck) {
                    // composite K chunk: shift k = ck>>1, channel base (ck&1)*32
                    int k   = ck >> 1;
                    int cb  = (ck & 1) * 32 + laneHi * 8;
                    int row = (l0 + lane15 + k + (LEN - 2)) & (LEN - 1);
                    // 16-bit A 16x32 layout: lanes<16 K={0..7,16..23}, lanes>=16 K={8..15,24..31}
                    v8h lo = *reinterpret_cast<const v8h*>(actIn + row * DIM + cb);
                    v8h hi = *reinterpret_cast<const v8h*>(actIn + row * DIM + cb + 16);
                    v16h a = __builtin_shufflevector(lo, hi,
                              0, 1, 2, 3, 4, 5, 6, 7, 8, 9, 10, 11, 12, 13, 14, 15);
                    acc = __builtin_amdgcn_wmma_f32_16x16x32_f16(
                              false, a, false, bfrag[ck], (short)0, acc, false, false);
                }
                // D layout: VGPR r -> row l0 + r + 8*laneHi, col = dch
                #pragma unroll
                for (int r = 0; r < 8; ++r) {
                    int rowl = l0 + r + laneHi * 8;
                    float y = acc[r] * s + sh;
                    actOut[rowl * DIM + dch] = (_Float16)elu1(y);
                }
            }
            __syncthreads();
        }

        // ---------------- Dense(1) head -> sDense ----------------
        if (tid < DIM) sScr[tid] = fcW[tid];
        __syncthreads();
        const float fcbv = fcb[0];
        for (int l = tid; l < LEN; l += 256) {
            float sacc = fcbv;
            #pragma unroll
            for (int d = 0; d < DIM; ++d)
                sacc = fmaf((float)sActA[l * DIM + d], sScr[d], sacc);
            sDense[l] = sacc;
        }
        // barrier at top of next branch iteration (or below) covers sDense
    }

    __syncthreads();
    const int* invRow = inv_perm + b * LEN;
    for (int l = tid; l < LEN; l += 256)
        out[b * LEN + l] = sDense[invRow[l]];
}

extern "C" void kernel_launch(void* const* d_in, const int* in_sizes, int n_in,
                              void* d_out, int out_size, void* d_ws, size_t ws_size,
                              hipStream_t stream) {
    (void)in_sizes; (void)n_in; (void)out_size; (void)d_ws; (void)ws_size;
    const float* dx       = (const float*)d_in[0];
    const float* de       = (const float*)d_in[1];
    const float* dI       = (const float*)d_in[2];
    const float* p        = (const float*)d_in[3];
    const float* Wx_first = (const float*)d_in[4];
    const float* bx_first = (const float*)d_in[5];
    const float* Wx_rest  = (const float*)d_in[6];
    const float* bx_rest  = (const float*)d_in[7];
    const float* gx       = (const float*)d_in[8];
    const float* btx      = (const float*)d_in[9];
    const float* mx       = (const float*)d_in[10];
    const float* vx       = (const float*)d_in[11];
    const float* WI_first = (const float*)d_in[12];
    const float* bI_first = (const float*)d_in[13];
    const float* WI_rest  = (const float*)d_in[14];
    const float* bI_rest  = (const float*)d_in[15];
    const float* gI       = (const float*)d_in[16];
    const float* btI      = (const float*)d_in[17];
    const float* mI       = (const float*)d_in[18];
    const float* vI       = (const float*)d_in[19];
    const float* fc1_W    = (const float*)d_in[20];
    const float* fc1_b    = (const float*)d_in[21];
    const float* fc2_W    = (const float*)d_in[22];
    const float* fc2_b    = (const float*)d_in[23];
    const int*   perm     = (const int*)d_in[24];
    const int*   inv_perm = (const int*)d_in[25];
    float* outp = (float*)d_out;

    fused_turbo_decoder<<<BATCH, 256, 0, stream>>>(
        dx, de, dI, p,
        Wx_first, bx_first, Wx_rest, bx_rest, gx, btx, mx, vx,
        WI_first, bI_first, WI_rest, bI_rest, gI, btI, mI, vI,
        fc1_W, fc1_b, fc2_W, fc2_b, perm, inv_perm, outp);
}
